// CfC_26551487823921
// MI455X (gfx1250) — matrix operand order repaired and checked by reading, not compiled, not run on hardware
//
#include <hip/hip_runtime.h>
#include <hip/hip_bf16.h>

typedef __attribute__((ext_vector_type(8)))  _Float16 v8h;
typedef __attribute__((ext_vector_type(16))) _Float16 v16h;
typedef __attribute__((ext_vector_type(8)))  float    v8f;
typedef __attribute__((ext_vector_type(4)))  float    v4f;
typedef __attribute__((ext_vector_type(4)))  unsigned int v4u;
typedef __attribute__((ext_vector_type(8)))  int      v8i;
typedef __attribute__((ext_vector_type(4)))  int      v4i;

#ifndef __has_builtin
#define __has_builtin(x) 0
#endif

#define B_  256
#define T_  512
#define D_  128
#define H_  256
#define U_  256
#define KZ  384            // D_ + H_
#define ZIN_STRIDE 392     // KZ + 8 halves pad (bank-conflict spread)
#define WBT_STRIDE 392
#define ZB_STRIDE  264     // U_ + 8

// workspace layout (bytes)
#define WS_WBT_OFF 0                              // _Float16 [256][392]  (Wb^T, [n][k])
#define WS_WHT_OFF (256 * WBT_STRIDE * 2)         // _Float16 [1024][256] (heads^T, [n][k])
#define WS_BB_OFF  (WS_WHT_OFF + 1024 * 256 * 2)  // float [256]
#define WS_HB_OFF  (WS_BB_OFF + 256 * 4)          // float [1024]

static __device__ inline v16h combine16(v8h lo, v8h hi) {
  v16h r;
#pragma unroll
  for (int i = 0; i < 8; ++i) { r[i] = lo[i]; r[8 + i] = hi[i]; }
  return r;
}

static __device__ inline v8f splat8(float v) {
  v8f r = {v, v, v, v, v, v, v, v};
  return r;
}

// ---------------------------------------------------------------------------
// Prep: convert weights to f16, transpose to [n][k] (WMMA B-fragment friendly)
// ---------------------------------------------------------------------------
__global__ __launch_bounds__(256) void cfc_prep(
    const float* __restrict__ Wb,  const float* __restrict__ bb,
    const float* __restrict__ Wff1, const float* __restrict__ bff1,
    const float* __restrict__ Wff2, const float* __restrict__ bff2,
    const float* __restrict__ Wta,  const float* __restrict__ bta,
    const float* __restrict__ Wtb,  const float* __restrict__ btb,
    _Float16* __restrict__ wbT, _Float16* __restrict__ whT,
    float* __restrict__ bbo, float* __restrict__ hbo) {
  int i = blockIdx.x * 256 + threadIdx.x;
  if (i < U_ * KZ) {                       // Wb is [KZ][U_] row-major
    int n = i / KZ, k = i % KZ;
    wbT[n * WBT_STRIDE + k] = (_Float16)Wb[k * U_ + n];
    return;
  }
  i -= U_ * KZ;
  if (i < 4 * H_ * U_) {                   // heads: r = hd*256 + n
    int r = i >> 8, k = i & 255;
    int hd = r >> 8, n = r & 255;
    const float* W = (hd == 0) ? Wff1 : (hd == 1) ? Wff2 : (hd == 2) ? Wta : Wtb;
    whT[(size_t)r * U_ + k] = (_Float16)W[k * H_ + n];
    return;
  }
  i -= 4 * H_ * U_;
  if (i < U_) { bbo[i] = bb[i]; return; }
  i -= U_;
  if (i < 4 * H_) {
    int hd = i >> 8, n = i & 255;
    const float* bv = (hd == 0) ? bff1 : (hd == 1) ? bff2 : (hd == 2) ? bta : btb;
    hbo[i] = bv[n];
  }
}

// ---------------------------------------------------------------------------
// Persistent recurrent scan: 16 workgroups, each owns a 16-row batch tile,
// h lives in LDS (as f16, inside the concat [x_t | h] staging tile).
// 8 waves x (2 backbone N-tiles + 2 hidden N-tiles * 4 heads) per step.
// Backbone weights arrive in LDS via the Tensor Data Mover (TDM).
// ---------------------------------------------------------------------------
__global__ __launch_bounds__(256) void cfc_scan(
    const float* __restrict__ x,
    const _Float16* __restrict__ wbT_g,
    const _Float16* __restrict__ whT_g,
    const float* __restrict__ bb_g,
    const float* __restrict__ hb_g,
    float* __restrict__ y, float* __restrict__ hT) {
  extern __shared__ __align__(16) char smem[];
  _Float16* wbT  = (_Float16*)smem;              // [256][WBT_STRIDE], LDS offset 0
  _Float16* zin  = wbT + 256 * WBT_STRIDE;       // [16][ZIN_STRIDE] = [x_t | h]
  _Float16* zbuf = zin + 16 * ZIN_STRIDE;        // [16][ZB_STRIDE]  = backbone act
  float*    bbs  = (float*)(zbuf + 16 * ZB_STRIDE);  // [256]
  float*    hbs  = bbs + 256;                        // [1024]

  const int tid   = threadIdx.x;
  const int wave  = tid >> 5;
  const int lane  = tid & 31;
  const int lq    = lane & 15;   // lane % 16
  const int hi    = lane >> 4;   // lane half
  const int bbase = blockIdx.x * 16;

  // ---- prologue: backbone weights -> LDS via TDM, biases -> LDS, h0 = 0 ----
#if __has_builtin(__builtin_amdgcn_tensor_load_to_lds)
  if (tid < 32) {  // one wave issues the DMA (EXEC is ignored by TDM)
    const unsigned long long ga = (unsigned long long)(size_t)wbT_g;
    const unsigned rowdw = (WBT_STRIDE * 2) / 4;   // 196 dwords per row
    const unsigned rows  = 256;
    v4u g0;
    g0[0] = 1u;                                          // count=1 (valid D#)
    g0[1] = 0u;                                          // lds_addr = 0 (wbT)
    g0[2] = (unsigned)(ga & 0xFFFFFFFFu);                // global_addr[31:0]
    g0[3] = (unsigned)((ga >> 32) & 0x01FFFFFFu)         // global_addr[56:32]
            | (2u << 30);                                // type = 2 ("image")
    v8i g1;
    g1[0] = (int)(2u << 16);                             // data_size = 4B, no mask
    g1[1] = (int)((rowdw & 0xFFFFu) << 16);              // tensor_dim0[15:0]
    g1[2] = (int)(((rowdw >> 16) & 0xFFFFu)              // tensor_dim0[31:16]
                  | ((rows & 0xFFFFu) << 16));           // tensor_dim1[15:0]
    g1[3] = (int)(((rows >> 16) & 0xFFFFu)               // tensor_dim1[31:16]
                  | ((rowdw & 0xFFFFu) << 16));          // tile_dim0 = 196
    g1[4] = (int)(rows & 0xFFFFu);                       // tile_dim1 = 256
    g1[5] = (int)rowdw;                                  // tensor_dim0_stride lo32
    g1[6] = 0;                                           // stride hi / dim1 stride
    g1[7] = 0;
    v4i gz4 = {0, 0, 0, 0};
    v8i gz8 = {0, 0, 0, 0, 0, 0, 0, 0};
    __builtin_amdgcn_tensor_load_to_lds(g0, g1, gz4, gz4, gz8, 0);
#if __has_builtin(__builtin_amdgcn_s_wait_tensorcnt)
    __builtin_amdgcn_s_wait_tensorcnt((short)0);
#else
    asm volatile("s_wait_tensorcnt 0x0" ::: "memory");
#endif
  }
#else
  {
    const uint4* s = (const uint4*)wbT_g;
    uint4* d = (uint4*)wbT;
    const int n16 = (256 * WBT_STRIDE * 2) / 16;
    for (int i = tid; i < n16; i += 256) d[i] = s[i];
  }
#endif
  {
    for (int i = tid; i < U_; i += 256) bbs[i] = bb_g[i];
    for (int i = tid; i < 4 * H_; i += 256) hbs[i] = hb_g[i];
    for (int i = tid; i < 16 * H_; i += 256) {
      int r = i >> 8, c = i & 255;
      zin[r * ZIN_STRIDE + D_ + c] = (_Float16)0.0f;
    }
    // warm L2 with the streamed head weights (512 KB, reread every step)
    const char* wp = (const char*)whT_g;
    for (int i = tid; i < (4 * H_ * U_ * 2) / 128; i += 256)
      __builtin_prefetch(wp + (size_t)i * 128, 0, 1);
  }
  __syncthreads();

  for (int t = 0; t < T_; ++t) {
    // ---- stage x_t (f32 -> f16) into zin[:, 0:128] ----
    {
      int e = tid * 8;              // 16*128 = 2048 elements, 8 per thread
      int r = e >> 7, d0 = e & 127;
      const float* xp = x + ((size_t)(bbase + r) * T_ + t) * D_ + d0;
      v4f f0 = *(const v4f*)xp;
      v4f f1 = *(const v4f*)(xp + 4);
      v8h o;
#pragma unroll
      for (int i = 0; i < 4; ++i) { o[i] = (_Float16)f0[i]; o[4 + i] = (_Float16)f1[i]; }
      *(v8h*)&zin[r * ZIN_STRIDE + d0] = o;
    }
    __syncthreads();

    // ---- backbone: z = lecun_tanh([x_t | h] @ Wb + bb) ----
    // two independent accumulator chains to pipeline the WMMA RAW latency
#pragma unroll
    for (int nn = 0; nn < 2; ++nn) {
      const int ncol = (wave * 2 + nn) * 16 + lq;
      v8f acc0 = splat8(bbs[ncol]);
      v8f acc1 = splat8(0.0f);
#pragma unroll
      for (int kk = 0; kk < KZ / 32; kk += 2) {
        v16h a0 = combine16(*(const v8h*)&zin[lq * ZIN_STRIDE + kk * 32 + hi * 8],
                            *(const v8h*)&zin[lq * ZIN_STRIDE + kk * 32 + hi * 8 + 16]);
        v16h b0 = combine16(*(const v8h*)&wbT[ncol * WBT_STRIDE + kk * 32 + hi * 8],
                            *(const v8h*)&wbT[ncol * WBT_STRIDE + kk * 32 + hi * 8 + 16]);
        v16h a1 = combine16(*(const v8h*)&zin[lq * ZIN_STRIDE + (kk + 1) * 32 + hi * 8],
                            *(const v8h*)&zin[lq * ZIN_STRIDE + (kk + 1) * 32 + hi * 8 + 16]);
        v16h b1 = combine16(*(const v8h*)&wbT[ncol * WBT_STRIDE + (kk + 1) * 32 + hi * 8],
                            *(const v8h*)&wbT[ncol * WBT_STRIDE + (kk + 1) * 32 + hi * 8 + 16]);
        acc0 = __builtin_amdgcn_wmma_f32_16x16x32_f16(false, a0, false, b0,
                                                      (short)0, acc0, false, false);
        acc1 = __builtin_amdgcn_wmma_f32_16x16x32_f16(false, a1, false, b1,
                                                      (short)0, acc1, false, false);
      }
#pragma unroll
      for (int i = 0; i < 8; ++i) {
        float zv = 1.7159f * tanhf(0.666f * (acc0[i] + acc1[i]));
        zbuf[(i + hi * 8) * ZB_STRIDE + ncol] = (_Float16)zv;
      }
    }
    __syncthreads();

    // ---- heads: ff1/ff2/ta/tb = z @ W* + b*, gated mix -> h_new ----
#pragma unroll
    for (int nn = 0; nn < 2; ++nn) {
      const int ncol = (wave * 2 + nn) * 16 + lq;
      v8f accs[4];
#pragma unroll
      for (int hd = 0; hd < 4; ++hd) accs[hd] = splat8(hbs[hd * H_ + ncol]);
#pragma unroll
      for (int kk = 0; kk < U_ / 32; ++kk) {
        v16h a = combine16(*(const v8h*)&zbuf[lq * ZB_STRIDE + kk * 32 + hi * 8],
                           *(const v8h*)&zbuf[lq * ZB_STRIDE + kk * 32 + hi * 8 + 16]);
#pragma unroll
        for (int hd = 0; hd < 4; ++hd) {
          const _Float16* wp =
              whT_g + ((size_t)(hd * H_ + ncol)) * U_ + kk * 32 + hi * 8;
          v16h b = combine16(*(const v8h*)wp, *(const v8h*)(wp + 16));
          accs[hd] = __builtin_amdgcn_wmma_f32_16x16x32_f16(false, a, false, b,
                                                            (short)0, accs[hd],
                                                            false, false);
        }
      }
#pragma unroll
      for (int i = 0; i < 8; ++i) {
        float f1 = tanhf(accs[0][i]);
        float f2 = tanhf(accs[1][i]);
        float s  = 1.0f / (1.0f + __expf(-(accs[2][i] + accs[3][i])));
        float hn = f1 + s * (f2 - f1);
        const int row  = i + hi * 8;
        const int brow = bbase + row;
        y[((size_t)brow * T_ + t) * H_ + ncol] = hn;
        zin[row * ZIN_STRIDE + D_ + ncol] = (_Float16)hn;   // h for next step
        if (t == T_ - 1) hT[(size_t)brow * H_ + ncol] = hn;
      }
    }
    __syncthreads();
  }
}

// ---------------------------------------------------------------------------
extern "C" void kernel_launch(void* const* d_in, const int* in_sizes, int n_in,
                              void* d_out, int out_size, void* d_ws, size_t ws_size,
                              hipStream_t stream) {
  (void)in_sizes; (void)n_in; (void)out_size; (void)ws_size;
  const float* x    = (const float*)d_in[0];
  const float* Wb   = (const float*)d_in[1];
  const float* bb   = (const float*)d_in[2];
  const float* Wff1 = (const float*)d_in[3];
  const float* bff1 = (const float*)d_in[4];
  const float* Wff2 = (const float*)d_in[5];
  const float* bff2 = (const float*)d_in[6];
  const float* Wta  = (const float*)d_in[7];
  const float* bta  = (const float*)d_in[8];
  const float* Wtb  = (const float*)d_in[9];
  const float* btb  = (const float*)d_in[10];

  char* ws = (char*)d_ws;
  _Float16* wbT = (_Float16*)(ws + WS_WBT_OFF);
  _Float16* whT = (_Float16*)(ws + WS_WHT_OFF);
  float*    bbo = (float*)(ws + WS_BB_OFF);
  float*    hbo = (float*)(ws + WS_HB_OFF);

  float* y  = (float*)d_out;                         // [B,T,H]
  float* hT = y + (size_t)B_ * T_ * H_;              // [B,H]

  const int prep_work = U_ * KZ + 4 * H_ * U_ + U_ + 4 * H_;
  cfc_prep<<<(prep_work + 255) / 256, 256, 0, stream>>>(
      Wb, bb, Wff1, bff1, Wff2, bff2, Wta, bta, Wtb, btb, wbT, whT, bbo, hbo);

  const size_t lds_bytes =
      (size_t)(256 * WBT_STRIDE + 16 * ZIN_STRIDE + 16 * ZB_STRIDE) * 2 +
      (256 + 1024) * 4;
  cfc_scan<<<B_ / 16, 256, lds_bytes, stream>>>(x, wbT, whT, bbo, hbo, y, hT);
}
